// Model_62148176773532
// MI455X (gfx1250) — matrix operand order repaired and checked by reading, not compiled, not run on hardware
//
#include <hip/hip_runtime.h>
#include <cstdint>
#include <cstddef>

#define DEVFN __device__ __forceinline__

typedef __attribute__((ext_vector_type(16))) __bf16 bf16x16;
typedef __attribute__((ext_vector_type(8)))  float  f32x8;

constexpr int N_NODES = 20000;
constexpr int N_EDGES = 512000;
constexpr int C = 32;
constexpr int L = 4;
constexpr float EPSF = 1e-5f;
constexpr int NC = N_NODES * C;           // 640000

// ---------------- workspace layout ----------------
// float region (bytes from ws base)
constexpr size_t OFF_F0   = 0;                              // [N][32] f32
constexpr size_t OFF_F1   = OFF_F0   + (size_t)NC * 4;      // 3 planes [x][N][32] f32
constexpr size_t OFF_AGG0 = OFF_F1   + (size_t)3 * NC * 4;  // [N][32] f32
constexpr size_t OFF_AGG1 = OFF_AGG0 + (size_t)NC * 4;      // 3 planes f32 (contiguous w/ AGG0)
constexpr size_t OFF_BF   = OFF_AGG1 + (size_t)3 * NC * 4;  // bf16 region (256B aligned: 8*NC*4)
// bf16 element offsets within bf16 region
constexpr size_t BO_F0H  = 0;                    // [N][32]
constexpr size_t BO_F1H  = BO_F0H + NC;          // 3 planes [x][N][32]
constexpr size_t BO_RW1T = BO_F1H + 3 * (size_t)NC;  // [L][32n][32k] (k padded 19->32)
constexpr size_t BO_RW2T = BO_RW1T + L * 1024;       // [L][128n][32k]
constexpr size_t BO_PW00 = BO_RW2T + L * 4096;
constexpr size_t BO_PW01 = BO_PW00 + L * 1024;
constexpr size_t BO_PW10 = BO_PW01 + L * 1024;
constexpr size_t BO_PW11 = BO_PW10 + L * 1024;
constexpr size_t BO_NW0  = BO_PW11 + L * 1024;
constexpr size_t BO_NW1  = BO_NW0  + L * 1024;
constexpr size_t BO_SW0  = BO_NW1  + L * 1024;
constexpr size_t BO_SW1  = BO_SW0  + L * 1024;
constexpr size_t BO_STN0 = BO_SW1  + L * 1024;   // [2][32][32]
constexpr size_t BO_STN1 = BO_STN0 + 2048;
constexpr size_t BO_STW0 = BO_STN1 + 2048;
constexpr size_t BO_STW1 = BO_STW0 + 2048;
constexpr size_t BO_ONW0 = BO_STW1 + 2048;
constexpr size_t BO_ONW1 = BO_ONW0 + 1024;
constexpr size_t BO_OW0P = BO_ONW1 + 1024;       // [16][32] (rows 8..15 zero)
constexpr size_t BO_OW1P = BO_OW0P + 512;

// ---------------- WMMA helpers ----------------
DEVFN f32x8 wmma_bf16(bf16x16 a, bf16x16 b, f32x8 c) {
  return __builtin_amdgcn_wmma_f32_16x16x32_bf16(false, a, false, b, (short)0, c,
                                                 false, false);
}

// A operand: row-major [16][32] bf16 tile.
// ISA 16-bit A layout: lane<16 -> row=lane, K {0..7}+{16..23};
//                      lane>=16 -> row=lane-16, K {8..15}+{24..31}.
DEVFN bf16x16 pack_A(const __bf16* T, int lane) {
  int m = lane & 15;
  int koff = (lane < 16) ? 0 : 8;
  const __bf16* row = T + m * 32;
  union { bf16x16 v; int4 q[2]; } u;
  u.q[0] = *(const int4*)(row + koff);
  u.q[1] = *(const int4*)(row + 16 + koff);
  return u.v;
}

// A operand gathered from global planes: row indices from LDS si[16].
DEVFN bf16x16 pack_rows(const __bf16* base, const int* si, int lane) {
  int m = lane & 15;
  int koff = (lane < 16) ? 0 : 8;
  const __bf16* row = base + (size_t)si[m] * 32;
  union { bf16x16 v; int4 q[2]; } u;
  u.q[0] = *(const int4*)(row + koff);
  u.q[1] = *(const int4*)(row + 16 + koff);
  return u.v;
}

// B operand: computes X @ W^T; W stored row-major [out][32]. B[k][n]=W[n][k].
// ISA 16-bit B layout: lane<16 -> n=lane, K {0..7}+{16..23}; lane>=16 -> n=lane-16, K {8..15}+{24..31}.
DEVFN bf16x16 pack_B(const __bf16* W, int ntile, int lane) {
  int n = (lane & 15) + (ntile << 4);
  const __bf16* row = W + n * 32 + ((lane < 16) ? 0 : 8);
  union { bf16x16 v; int4 q[2]; } u;
  u.q[0] = *(const int4*)(row);
  u.q[1] = *(const int4*)(row + 16);
  return u.v;
}

DEVFN float eluf(float x) { return x > 0.f ? x : (__expf(x) - 1.f); }

// Native L2-side fp32 atomic add, device scope (ISA GLOBAL_ATOMIC_ADD_F32).
// Guarantees no CAS-loop expansion for the segment-sum scatter.
DEVFN void atomAddF(float* p, float v) {
  asm volatile("global_atomic_add_f32 %0, %1, off scope:SCOPE_DEV"
               :: "v"(p), "v"(v) : "memory");
}

// ---------------- weight prep (fp32 -> bf16, transposes, padding) ----------------
__global__ void prep_weights(
    const float* rW1, const float* rW2,
    const float* pW00, const float* pW01, const float* pW10, const float* pW11,
    const float* nW0, const float* nW1, const float* sW0, const float* sW1,
    const float* stnW0, const float* stnW1, const float* stW0, const float* stW1,
    const float* onW0, const float* onW1, const float* oW0, const float* oW1,
    __bf16* wb)
{
  int t = threadIdx.x;
  // rW1T[l][n][k] = rW1[l][k][n], k<19 else 0
  for (int i = t; i < L * 1024; i += 256) {
    int l = i >> 10, r = i & 1023, n = r >> 5, k = r & 31;
    float v = (k < 19) ? rW1[l * 19 * 32 + k * 32 + n] : 0.f;
    wb[BO_RW1T + i] = (__bf16)v;
  }
  // rW2T[l][n][k] = rW2[l][k][n], n<128
  for (int i = t; i < L * 4096; i += 256) {
    int l = i >> 12, r = i & 4095, n = r >> 5, k = r & 31;
    wb[BO_RW2T + i] = (__bf16)rW2[l * 4096 + k * 128 + n];
  }
  auto cvt = [&](const float* src, size_t dst, int cnt) {
    for (int i = t; i < cnt; i += 256) wb[dst + i] = (__bf16)src[i];
  };
  cvt(pW00, BO_PW00, L * 1024);  cvt(pW01, BO_PW01, L * 1024);
  cvt(pW10, BO_PW10, L * 1024);  cvt(pW11, BO_PW11, L * 1024);
  cvt(nW0,  BO_NW0,  L * 1024);  cvt(nW1,  BO_NW1,  L * 1024);
  cvt(sW0,  BO_SW0,  L * 1024);  cvt(sW1,  BO_SW1,  L * 1024);
  cvt(stnW0, BO_STN0, 2048);     cvt(stnW1, BO_STN1, 2048);
  cvt(stW0,  BO_STW0, 2048);     cvt(stW1,  BO_STW1, 2048);
  cvt(onW0,  BO_ONW0, 1024);     cvt(onW1,  BO_ONW1, 1024);
  for (int i = t; i < 512; i += 256) {
    int n = i >> 5, k = i & 31;
    wb[BO_OW0P + i] = (__bf16)((n < 8) ? oW0[n * 32 + k] : 0.f);
    wb[BO_OW1P + i] = (__bf16)((n < 8) ? oW1[n * 32 + k] : 0.f);
  }
}

// ---------------- node init: f0 = embed[atm] @ W_in0^T ; f1 = 0 ----------------
__global__ void init_nodes(const int* __restrict__ atm, const float* __restrict__ embed,
                           const float* __restrict__ Win0,
                           float* __restrict__ f0, float* __restrict__ f1,
                           __bf16* __restrict__ f0h, __bf16* __restrict__ f1h)
{
  int idx = blockIdx.x * 256 + threadIdx.x;
  if (idx >= NC) return;
  int n = idx >> 5, c = idx & 31;
  const float* em = embed + atm[n] * 16;
  const float* w  = Win0 + c * 16;
  float acc = 0.f;
#pragma unroll
  for (int k = 0; k < 16; ++k) acc += em[k] * w[k];
  f0[idx] = acc;
  f0h[idx] = (__bf16)acc;
#pragma unroll
  for (int x = 0; x < 3; ++x) {
    f1[(size_t)x * NC + idx] = 0.f;
    f1h[(size_t)x * NC + idx] = (__bf16)0.f;
  }
}

// ---------------- fused edge kernel: RBF -> MLP -> messages -> atomic scatter ----------------
// LDS weight cache offsets (bf16 elements)
constexpr int SW_RW1T = 0;
constexpr int SW_RW2T = SW_RW1T + 1024;
constexpr int SW_P00  = SW_RW2T + 4096;
constexpr int SW_P01  = SW_P00 + 1024;
constexpr int SW_P10  = SW_P01 + 1024;
constexpr int SW_P11  = SW_P10 + 1024;
constexpr int SW_TOT  = SW_P11 + 1024;   // 9216 bf16 = 18KB

__launch_bounds__(128)
__global__ void edge_kernel(
    const float* __restrict__ pos, const int* __restrict__ esrc, const int* __restrict__ edst,
    const float* __restrict__ efeat,
    const __bf16* __restrict__ f0h, const __bf16* __restrict__ f1h,
    const __bf16* __restrict__ rW1T, const __bf16* __restrict__ rW2T,
    const float* __restrict__ rb1, const float* __restrict__ rb2,
    const __bf16* __restrict__ pW00, const __bf16* __restrict__ pW01,
    const __bf16* __restrict__ pW10, const __bf16* __restrict__ pW11,
    float* __restrict__ agg0, float* __restrict__ agg1)
{
  __shared__ __align__(16) __bf16 sB1[4][512];     // edge_in, later proj
  __shared__ __align__(16) __bf16 sB2[4][512];     // hidden h
  __shared__ __align__(16) float  sRad[4][2048];   // [16][128] radial outputs
  __shared__ __align__(16) __bf16 sWgt[SW_TOT];    // block-shared layer weights
  __shared__ float sRh[4][16][3];
  __shared__ int   sSi[4][16], sDi[4][16];

  int w = threadIdx.x >> 5, lane = threadIdx.x & 31;
  long base = ((long)blockIdx.x * 4 + w) * 16;

  // Cooperative weight staging into LDS (all 128 threads).
  {
    auto cp = [&](int dstOff, const __bf16* src, int nElems) {
      int4* d4 = (int4*)(sWgt + dstOff);
      const int4* s4 = (const int4*)src;
      for (int i = threadIdx.x; i < (nElems >> 3); i += 128) d4[i] = s4[i];
    };
    cp(SW_RW1T, rW1T, 1024);
    cp(SW_RW2T, rW2T, 4096);
    cp(SW_P00,  pW00, 1024);
    cp(SW_P01,  pW01, 1024);
    cp(SW_P10,  pW10, 1024);
    cp(SW_P11,  pW11, 1024);
  }

  // S0: geometry + RBF + edge features (lanes 0..15, one edge each)
  if (lane < 16) {
    int m = lane;
    long e = base + m;
    int s = esrc[e], d = edst[e];
    sSi[w][m] = s; sDi[w][m] = d;
    // Prefetch the gathered node-feature rows (consumed in S3/S4, two WMMA
    // stages later) so the random-access L2 fills overlap the edge MLP.
    __builtin_prefetch(f0h + (size_t)s * 32, 0, 3);
    __builtin_prefetch(f1h + (size_t)s * 32, 0, 3);
    __builtin_prefetch(f1h + (size_t)NC + (size_t)s * 32, 0, 3);
    __builtin_prefetch(f1h + 2 * (size_t)NC + (size_t)s * 32, 0, 3);
    float rx = pos[d * 3 + 0] - pos[s * 3 + 0];
    float ry = pos[d * 3 + 1] - pos[s * 3 + 1];
    float rz = pos[d * 3 + 2] - pos[s * 3 + 2];
    float dist = sqrtf(rx * rx + ry * ry + rz * rz + EPSF);
    float inv = 1.f / dist;
    sRh[w][m][0] = rx * inv; sRh[w][m][1] = ry * inv; sRh[w][m][2] = rz * inv;
    __bf16* ein = sB1[w] + m * 32;
#pragma unroll
    for (int j = 0; j < 16; ++j) {
      float u = (dist - (float)j * (10.f / 15.f)) * (1.f / 0.75f);
      ein[j] = (__bf16)__expf(-u * u);
    }
    ein[16] = (__bf16)efeat[e * 3 + 0];
    ein[17] = (__bf16)efeat[e * 3 + 1];
    ein[18] = (__bf16)efeat[e * 3 + 2];
#pragma unroll
    for (int j = 19; j < 32; ++j) ein[j] = (__bf16)0.f;
  }
  __syncthreads();

  f32x8 z = {};
  int col0 = lane & 15;
  int rb   = (lane < 16) ? 0 : 8;

  // S1: h = elu(edge_in @ rW1 + rb1)
  {
    bf16x16 a = pack_A(sB1[w], lane);
#pragma unroll
    for (int nt = 0; nt < 2; ++nt) {
      f32x8 d = wmma_bf16(a, pack_B(sWgt + SW_RW1T, nt, lane), z);
      int col = col0 + nt * 16;
      float b = rb1[col];
#pragma unroll
      for (int r = 0; r < 8; ++r) sB2[w][(rb + r) * 32 + col] = (__bf16)eluf(d[r] + b);
    }
  }
  __syncthreads();

  // S2: rad = h @ rW2 + rb2   ([16][128] in LDS)
  {
    bf16x16 a = pack_A(sB2[w], lane);
#pragma unroll
    for (int nt = 0; nt < 8; ++nt) {
      f32x8 d = wmma_bf16(a, pack_B(sWgt + SW_RW2T, nt, lane), z);
      int col = col0 + nt * 16;
      float b = rb2[col];
#pragma unroll
      for (int r = 0; r < 8; ++r) sRad[w][(rb + r) * 128 + col] = d[r] + b;
    }
  }
  __syncthreads();

  // S3: proj[e][c] = f1s[e][c][:] . rhat[e]  -> sB1 (bf16)
  {
    int m = lane & 15;
    int cb = (lane < 16) ? 0 : 16;
    int s = sSi[w][m];
    float hx = sRh[w][m][0], hy = sRh[w][m][1], hz = sRh[w][m][2];
    const __bf16* p0 = f1h + (size_t)s * 32;
#pragma unroll
    for (int i = 0; i < 16; ++i) {
      int c = cb + i;
      float v = (float)p0[c] * hx + (float)p0[(size_t)NC + c] * hy +
                (float)p0[2 * (size_t)NC + c] * hz;
      sB1[w][m * 32 + c] = (__bf16)v;
    }
  }
  __syncthreads();

  // S4: six channel-mixing matmuls + modulation + atomic scatter
  {
    bf16x16 aF0 = pack_rows(f0h, sSi[w], lane);
    bf16x16 aPr = pack_A(sB1[w], lane);
    bf16x16 aFx = pack_rows(f1h, sSi[w], lane);
    bf16x16 aFy = pack_rows(f1h + (size_t)NC, sSi[w], lane);
    bf16x16 aFz = pack_rows(f1h + 2 * (size_t)NC, sSi[w], lane);
#pragma unroll
    for (int nt = 0; nt < 2; ++nt) {
      f32x8 d00 = wmma_bf16(aF0, pack_B(sWgt + SW_P00, nt, lane), z);
      f32x8 d01 = wmma_bf16(aF0, pack_B(sWgt + SW_P01, nt, lane), z);
      f32x8 d10 = wmma_bf16(aPr, pack_B(sWgt + SW_P10, nt, lane), z);
      f32x8 dxx = wmma_bf16(aFx, pack_B(sWgt + SW_P11, nt, lane), z);
      f32x8 dyy = wmma_bf16(aFy, pack_B(sWgt + SW_P11, nt, lane), z);
      f32x8 dzz = wmma_bf16(aFz, pack_B(sWgt + SW_P11, nt, lane), z);
      int col = col0 + nt * 16;
#pragma unroll
      for (int r = 0; r < 8; ++r) {
        int m = rb + r;
        float r00 = sRad[w][m * 128 + col];
        float r01 = sRad[w][m * 128 + 32 + col];
        float r10 = sRad[w][m * 128 + 64 + col];
        float r11 = sRad[w][m * 128 + 96 + col];
        float hx = sRh[w][m][0], hy = sRh[w][m][1], hz = sRh[w][m][2];
        int dd = sDi[w][m];
        float m0 = r00 * d00[r] + r10 * d10[r];
        float s1 = r01 * d01[r];
        size_t g = (size_t)dd * 32 + col;
        atomAddF(agg0 + g, m0);
        atomAddF(agg1 + g, s1 * hx + r11 * dxx[r]);
        atomAddF(agg1 + (size_t)NC + g, s1 * hy + r11 * dyy[r]);
        atomAddF(agg1 + 2 * (size_t)NC + g, s1 * hz + r11 * dzz[r]);
      }
    }
  }
}

// ---------------- node kernel: residual + NormSE3 + self-interaction ----------------
__launch_bounds__(64)
__global__ void node_kernel(
    float* __restrict__ f0, float* __restrict__ f1,
    const float* __restrict__ agg0, const float* __restrict__ agg1,
    __bf16* __restrict__ f0h, __bf16* __restrict__ f1h,
    const __bf16* __restrict__ nW0, const __bf16* __restrict__ nW1,
    const __bf16* __restrict__ sW0, const __bf16* __restrict__ sW1)
{
  __shared__ __align__(16) __bf16 sA0[2][512];
  __shared__ __align__(16) __bf16 sA1[2][512];
  __shared__ float sP0[2][512];
  __shared__ float sP1[2][3][512];

  int w = threadIdx.x >> 5, lane = threadIdx.x & 31;
  int base = (blockIdx.x * 2 + w) * 16;
  int m = lane & 15;
  int cb = (lane < 16) ? 0 : 16;
  int col0 = lane & 15;
  int rbr = (lane < 16) ? 0 : 8;

  float rn0[16], rn1[16];
  float s0 = 0, q0 = 0, s1 = 0, q1 = 0;
  size_t nb = (size_t)(base + m) * 32;
#pragma unroll
  for (int i = 0; i < 16; ++i) {
    int c = cb + i;
    size_t g = nb + c;
    float h0 = f0[g] + (agg0 ? agg0[g] : 0.f);
    float a = f1[g] + (agg1 ? agg1[g] : 0.f);
    float b = f1[(size_t)NC + g] + (agg1 ? agg1[(size_t)NC + g] : 0.f);
    float d = f1[2 * (size_t)NC + g] + (agg1 ? agg1[2 * (size_t)NC + g] : 0.f);
    float n0v = sqrtf(h0 * h0 + EPSF);
    float n1v = sqrtf(a * a + b * b + d * d + EPSF);
    sP0[w][m * 32 + c] = h0 / n0v;
    sP1[w][0][m * 32 + c] = a / n1v;
    sP1[w][1][m * 32 + c] = b / n1v;
    sP1[w][2][m * 32 + c] = d / n1v;
    rn0[i] = n0v; rn1[i] = n1v;
    s0 += n0v; q0 += n0v * n0v; s1 += n1v; q1 += n1v * n1v;
  }
  s0 += __shfl_xor(s0, 16, 32); q0 += __shfl_xor(q0, 16, 32);
  s1 += __shfl_xor(s1, 16, 32); q1 += __shfl_xor(q1, 16, 32);
  float mu0 = s0 * (1.f / 32.f), var0 = q0 * (1.f / 32.f) - mu0 * mu0;
  float mu1 = s1 * (1.f / 32.f), var1 = q1 * (1.f / 32.f) - mu1 * mu1;
  float iv0 = rsqrtf(var0 + 1e-5f), iv1 = rsqrtf(var1 + 1e-5f);
#pragma unroll
  for (int i = 0; i < 16; ++i) {
    sA0[w][m * 32 + cb + i] = (__bf16)((rn0[i] - mu0) * iv0);
    sA1[w][m * 32 + cb + i] = (__bf16)((rn1[i] - mu1) * iv1);
  }
  __syncthreads();

  f32x8 z = {};
  bf16x16 a0 = pack_A(sA0[w], lane);
  bf16x16 a1 = pack_A(sA1[w], lane);
  f32x8 d0[2], e1[2];
#pragma unroll
  for (int nt = 0; nt < 2; ++nt) {
    d0[nt] = wmma_bf16(a0, pack_B(nW0, nt, lane), z);
    e1[nt] = wmma_bf16(a1, pack_B(nW1, nt, lane), z);
  }
  __syncthreads();
#pragma unroll
  for (int nt = 0; nt < 2; ++nt) {
    int col = col0 + nt * 16;
#pragma unroll
    for (int r = 0; r < 8; ++r) {
      int mm = rbr + r;
      sA0[w][mm * 32 + col] = (__bf16)(eluf(d0[nt][r]) * sP0[w][mm * 32 + col]);
      e1[nt][r] = eluf(e1[nt][r]);
    }
  }
  __syncthreads();
  // f0' = o0 @ sW0^T
  {
    bf16x16 ao = pack_A(sA0[w], lane);
#pragma unroll
    for (int nt = 0; nt < 2; ++nt) {
      f32x8 d = wmma_bf16(ao, pack_B(sW0, nt, lane), z);
      int col = col0 + nt * 16;
#pragma unroll
      for (int r = 0; r < 8; ++r) {
        size_t g = (size_t)(base + rbr + r) * 32 + col;
        f0[g] = d[r];
        f0h[g] = (__bf16)d[r];
      }
    }
  }
  // f1'_x = (coef * p1_x) @ sW1^T
  for (int x = 0; x < 3; ++x) {
    __syncthreads();
#pragma unroll
    for (int nt = 0; nt < 2; ++nt) {
      int col = col0 + nt * 16;
#pragma unroll
      for (int r = 0; r < 8; ++r) {
        int mm = rbr + r;
        sA1[w][mm * 32 + col] = (__bf16)(e1[nt][r] * sP1[w][x][mm * 32 + col]);
      }
    }
    __syncthreads();
    bf16x16 av = pack_A(sA1[w], lane);
#pragma unroll
    for (int nt = 0; nt < 2; ++nt) {
      f32x8 d = wmma_bf16(av, pack_B(sW1, nt, lane), z);
      int col = col0 + nt * 16;
#pragma unroll
      for (int r = 0; r < 8; ++r) {
        size_t g = (size_t)x * NC + (size_t)(base + rbr + r) * 32 + col;
        f1[g] = d[r];
        f1h[g] = (__bf16)d[r];
      }
    }
  }
}

// ---------------- final kernel: NormSE3(onW) + output projection ----------------
__launch_bounds__(64)
__global__ void final_kernel(
    const float* __restrict__ f0, const float* __restrict__ f1,
    const __bf16* __restrict__ onW0, const __bf16* __restrict__ onW1,
    const __bf16* __restrict__ oW0p, const __bf16* __restrict__ oW1p,
    float* __restrict__ out)
{
  __shared__ __align__(16) __bf16 sA0[2][512];
  __shared__ __align__(16) __bf16 sA1[2][512];
  __shared__ float sP0[2][512];
  __shared__ float sP1[2][3][512];

  int w = threadIdx.x >> 5, lane = threadIdx.x & 31;
  int base = (blockIdx.x * 2 + w) * 16;
  int m = lane & 15;
  int cb = (lane < 16) ? 0 : 16;
  int col0 = lane & 15;
  int rbr = (lane < 16) ? 0 : 8;

  float rn0[16], rn1[16];
  float s0 = 0, q0 = 0, s1 = 0, q1 = 0;
  size_t nb = (size_t)(base + m) * 32;
#pragma unroll
  for (int i = 0; i < 16; ++i) {
    int c = cb + i;
    size_t g = nb + c;
    float h0 = f0[g];
    float a = f1[g];
    float b = f1[(size_t)NC + g];
    float d = f1[2 * (size_t)NC + g];
    float n0v = sqrtf(h0 * h0 + EPSF);
    float n1v = sqrtf(a * a + b * b + d * d + EPSF);
    sP0[w][m * 32 + c] = h0 / n0v;
    sP1[w][0][m * 32 + c] = a / n1v;
    sP1[w][1][m * 32 + c] = b / n1v;
    sP1[w][2][m * 32 + c] = d / n1v;
    rn0[i] = n0v; rn1[i] = n1v;
    s0 += n0v; q0 += n0v * n0v; s1 += n1v; q1 += n1v * n1v;
  }
  s0 += __shfl_xor(s0, 16, 32); q0 += __shfl_xor(q0, 16, 32);
  s1 += __shfl_xor(s1, 16, 32); q1 += __shfl_xor(q1, 16, 32);
  float mu0 = s0 * (1.f / 32.f), var0 = q0 * (1.f / 32.f) - mu0 * mu0;
  float mu1 = s1 * (1.f / 32.f), var1 = q1 * (1.f / 32.f) - mu1 * mu1;
  float iv0 = rsqrtf(var0 + 1e-5f), iv1 = rsqrtf(var1 + 1e-5f);
#pragma unroll
  for (int i = 0; i < 16; ++i) {
    sA0[w][m * 32 + cb + i] = (__bf16)((rn0[i] - mu0) * iv0);
    sA1[w][m * 32 + cb + i] = (__bf16)((rn1[i] - mu1) * iv1);
  }
  __syncthreads();

  f32x8 z = {};
  bf16x16 a0 = pack_A(sA0[w], lane);
  bf16x16 a1 = pack_A(sA1[w], lane);
  f32x8 d0[2], e1[2];
#pragma unroll
  for (int nt = 0; nt < 2; ++nt) {
    d0[nt] = wmma_bf16(a0, pack_B(onW0, nt, lane), z);
    e1[nt] = wmma_bf16(a1, pack_B(onW1, nt, lane), z);
  }
  __syncthreads();
#pragma unroll
  for (int nt = 0; nt < 2; ++nt) {
    int col = col0 + nt * 16;
#pragma unroll
    for (int r = 0; r < 8; ++r) {
      int mm = rbr + r;
      sA0[w][mm * 32 + col] = (__bf16)(eluf(d0[nt][r]) * sP0[w][mm * 32 + col]);
      e1[nt][r] = eluf(e1[nt][r]);
    }
  }
  __syncthreads();
  // o0 = n0 @ oW0^T (only 8 output cols valid)
  {
    bf16x16 ao = pack_A(sA0[w], lane);
    f32x8 d = wmma_bf16(ao, pack_B(oW0p, 0, lane), z);
    if (col0 < 8) {
#pragma unroll
      for (int r = 0; r < 8; ++r)
        out[(size_t)(base + rbr + r) * 32 + col0] = d[r];
    }
  }
  // o1_x = (coef * p1_x) @ oW1^T ; out layout: 8 + c*3 + x
  for (int x = 0; x < 3; ++x) {
    __syncthreads();
#pragma unroll
    for (int nt = 0; nt < 2; ++nt) {
      int col = col0 + nt * 16;
#pragma unroll
      for (int r = 0; r < 8; ++r) {
        int mm = rbr + r;
        sA1[w][mm * 32 + col] = (__bf16)(e1[nt][r] * sP1[w][x][mm * 32 + col]);
      }
    }
    __syncthreads();
    bf16x16 av = pack_A(sA1[w], lane);
    f32x8 d = wmma_bf16(av, pack_B(oW1p, 0, lane), z);
    if (col0 < 8) {
#pragma unroll
      for (int r = 0; r < 8; ++r)
        out[(size_t)(base + rbr + r) * 32 + 8 + col0 * 3 + x] = d[r];
    }
  }
}

// ---------------- host launcher ----------------
extern "C" void kernel_launch(void* const* d_in, const int* in_sizes, int n_in,
                              void* d_out, int out_size, void* d_ws, size_t ws_size,
                              hipStream_t stream) {
  (void)in_sizes; (void)n_in; (void)out_size; (void)ws_size;
  const float* pos   = (const float*)d_in[0];
  const int*   atm   = (const int*)d_in[1];
  const int*   esrc  = (const int*)d_in[2];
  const int*   edst  = (const int*)d_in[3];
  const float* efeat = (const float*)d_in[4];
  const float* embed = (const float*)d_in[5];
  const float* Win0  = (const float*)d_in[6];
  const float* rW1   = (const float*)d_in[7];
  const float* rb1   = (const float*)d_in[8];
  const float* rW2   = (const float*)d_in[9];
  const float* rb2   = (const float*)d_in[10];
  const float* pW00  = (const float*)d_in[11];
  const float* pW01  = (const float*)d_in[12];
  const float* pW10  = (const float*)d_in[13];
  const float* pW11  = (const float*)d_in[14];
  const float* nW0   = (const float*)d_in[15];
  const float* nW1   = (const float*)d_in[16];
  const float* sW0   = (const float*)d_in[17];
  const float* sW1   = (const float*)d_in[18];
  const float* stnW0 = (const float*)d_in[19];
  const float* stnW1 = (const float*)d_in[20];
  const float* stW0  = (const float*)d_in[21];
  const float* stW1  = (const float*)d_in[22];
  const float* onW0  = (const float*)d_in[23];
  const float* onW1  = (const float*)d_in[24];
  const float* oW0   = (const float*)d_in[25];
  const float* oW1   = (const float*)d_in[26];

  char* ws = (char*)d_ws;
  float* f0   = (float*)(ws + OFF_F0);
  float* f1   = (float*)(ws + OFF_F1);
  float* agg0 = (float*)(ws + OFF_AGG0);
  float* agg1 = (float*)(ws + OFF_AGG1);
  __bf16* wb  = (__bf16*)(ws + OFF_BF);
  __bf16* f0h = wb + BO_F0H;
  __bf16* f1h = wb + BO_F1H;

  prep_weights<<<1, 256, 0, stream>>>(rW1, rW2, pW00, pW01, pW10, pW11,
                                      nW0, nW1, sW0, sW1, stnW0, stnW1, stW0, stW1,
                                      onW0, onW1, oW0, oW1, wb);
  init_nodes<<<(NC + 255) / 256, 256, 0, stream>>>(atm, embed, Win0, f0, f1, f0h, f1h);

  for (int l = 0; l < L; ++l) {
    hipMemsetAsync(agg0, 0, (size_t)4 * NC * sizeof(float), stream);  // agg0+agg1 contiguous
    edge_kernel<<<N_EDGES / 64, 128, 0, stream>>>(
        pos, esrc, edst, efeat, f0h, f1h,
        wb + BO_RW1T + (size_t)l * 1024, wb + BO_RW2T + (size_t)l * 4096,
        rb1 + l * 32, rb2 + l * 128,
        wb + BO_PW00 + (size_t)l * 1024, wb + BO_PW01 + (size_t)l * 1024,
        wb + BO_PW10 + (size_t)l * 1024, wb + BO_PW11 + (size_t)l * 1024,
        agg0, agg1);
    node_kernel<<<N_NODES / 32, 64, 0, stream>>>(
        f0, f1, agg0, agg1, f0h, f1h,
        wb + BO_NW0 + (size_t)l * 1024, wb + BO_NW1 + (size_t)l * 1024,
        wb + BO_SW0 + (size_t)l * 1024, wb + BO_SW1 + (size_t)l * 1024);
  }
  for (int i = 0; i < 2; ++i) {
    node_kernel<<<N_NODES / 32, 64, 0, stream>>>(
        f0, f1, nullptr, nullptr, f0h, f1h,
        wb + BO_STN0 + (size_t)i * 1024, wb + BO_STN1 + (size_t)i * 1024,
        wb + BO_STW0 + (size_t)i * 1024, wb + BO_STW1 + (size_t)i * 1024);
  }
  final_kernel<<<N_NODES / 32, 64, 0, stream>>>(f0, f1, wb + BO_ONW0, wb + BO_ONW1,
                                                wb + BO_OW0P, wb + BO_OW1P,
                                                (float*)d_out);
}
